// EmbTransformer_59030030516362
// MI455X (gfx1250) — matrix operand order repaired and checked by reading, not compiled, not run on hardware
//
#include <hip/hip_runtime.h>

// ---------------------------------------------------------------------------
// GNN message passing + linear:  out = segment_sum(src_h[src_idx], dst_idx) @ W^T + b
//   src_h: [N,512] f32, W: [512,512] f32, b: [512] f32, idx: [E] i32
// Phase 0: zero agg (in d_ws)
// Phase 1: edge scatter-add with f32 global atomics (agg is L2-resident: 102MB < 192MB)
// Phase 2: WMMA GEMM, f16 hi/lo 3-pass split for ~fp32 accuracy on the matrix pipes
// ---------------------------------------------------------------------------

#define D_K 512
#define TILE_M 128          // 8 waves x 16 rows
#define TILE_N 64           // 4 x 16 cols per wave
#define LDS_STRIDE 520      // 512 + 8 halves padding -> conflict-free ds_load_b128

typedef __attribute__((ext_vector_type(16))) _Float16 v16h;
typedef __attribute__((ext_vector_type(8)))  _Float16 h8;
typedef __attribute__((ext_vector_type(4)))  _Float16 h4;
typedef __attribute__((ext_vector_type(8)))  float    v8f;

// ---------------- Phase 0: zero the aggregation buffer ----------------------
__global__ __launch_bounds__(256) void gnn_zero_kernel(float4* __restrict__ p, int n4) {
  int i = blockIdx.x * 256 + threadIdx.x;
  if (i < n4) p[i] = float4{0.f, 0.f, 0.f, 0.f};
}

// ---------------- Phase 1: gather + scatter-add (atomics into L2) -----------
__global__ __launch_bounds__(256) void gnn_scatter_kernel(
    const float* __restrict__ src_h,
    const int*   __restrict__ src_idx,
    const int*   __restrict__ dst_idx,
    float*       __restrict__ agg,
    int n_edges)
{
  int tid = blockIdx.x * 256 + threadIdx.x;
  int e = tid >> 7;                 // 128 float4-chunks cover 512 floats
  if (e >= n_edges) return;
  int c = (tid & 127) << 2;         // float offset within the row
  int s = src_idx[e];
  int d = dst_idx[e];
  float4 v = *(const float4*)(src_h + ((size_t)s << 9) + c);
  float* p = agg + ((size_t)d << 9) + c;
  // hardware f32 atomic add (global_atomic_add_f32), no return -> STOREcnt path
  unsafeAtomicAdd(p + 0, v.x);
  unsafeAtomicAdd(p + 1, v.y);
  unsafeAtomicAdd(p + 2, v.z);
  unsafeAtomicAdd(p + 3, v.w);
}

// ---------------- Phase 2: WMMA GEMM  out[M,512] = agg[M,512] @ W[512,512]^T + b
__global__ __launch_bounds__(256) void gnn_gemm_kernel(
    const float* __restrict__ agg,
    const float* __restrict__ W,
    const float* __restrict__ b,
    float*       __restrict__ out,
    int M)
{
  extern __shared__ _Float16 smem[];
  _Float16* Wh = smem;                          // hi halves of W tile
  _Float16* Wl = smem + TILE_N * LDS_STRIDE;    // lo halves of W tile

  const int nbase = blockIdx.y * TILE_N;
  const int m0    = blockIdx.x * TILE_M;

  // Cooperative convert of W[nbase:nbase+64, 0:512] f32 -> f16 hi/lo in LDS.
  for (int c = threadIdx.x; c < TILE_N * (D_K / 4); c += 256) {
    int row = c >> 7;                 // 128 float4 chunks per row
    int k   = (c & 127) << 2;
    float4 w = *(const float4*)(W + ((size_t)(nbase + row) << 9) + k);
    float wf[4] = {w.x, w.y, w.z, w.w};
    h4 hh, hl;
#pragma unroll
    for (int i = 0; i < 4; ++i) {
      _Float16 h = (_Float16)wf[i];
      hh[i] = h;
      hl[i] = (_Float16)(wf[i] - (float)h);   // residual, exactly representable
    }
    *(h4*)(Wh + row * LDS_STRIDE + k) = hh;
    *(h4*)(Wl + row * LDS_STRIDE + k) = hl;
  }
  __syncthreads();

  const int lane = threadIdx.x & 31;
  const int wv   = threadIdx.x >> 5;   // wave id 0..7 -> 16-row slice
  const int lr   = lane & 15;
  const int hi   = lane >> 4;          // lane half selects K sub-block per ISA layout

  int rowA = m0 + wv * 16 + lr;
  if (rowA >= M) rowA = M - 1;         // clamped load; stores are guarded below
  const float* arow = agg + ((size_t)rowA << 9);

  v8f acc[4];
#pragma unroll
  for (int j = 0; j < 4; ++j) {
    float bv = b[nbase + j * 16 + lr];
#pragma unroll
    for (int r = 0; r < 8; ++r) acc[j][r] = bv;   // seed accumulator with bias
  }

  for (int k0 = 0; k0 < D_K; k0 += 32) {
    // ---- A fragment (16x32 f16): ISA layout
    //   lanes 0-15 : K = {k0+0..7,  k0+16..23}
    //   lanes 16-31: K = {k0+8..15, k0+24..31}
    const float* ap = arow + k0 + hi * 8;
    float4 x0 = *(const float4*)(ap);
    float4 x1 = *(const float4*)(ap + 4);
    float4 x2 = *(const float4*)(ap + 16);
    float4 x3 = *(const float4*)(ap + 20);
    float xs[16] = {x0.x, x0.y, x0.z, x0.w, x1.x, x1.y, x1.z, x1.w,
                    x2.x, x2.y, x2.z, x2.w, x3.x, x3.y, x3.z, x3.w};
    v16h Ah, Al;
#pragma unroll
    for (int i = 0; i < 16; ++i) {
      _Float16 h = (_Float16)xs[i];
      Ah[i] = h;
      Al[i] = (_Float16)(xs[i] - (float)h);
    }

#pragma unroll
    for (int j = 0; j < 4; ++j) {
      // ---- B fragment (32x16 f16): lane lr = column, K layout matches A (lanes 16-31 hold K+8 / K+24)
      int rb = (j * 16 + lr) * LDS_STRIDE + k0 + hi * 8;
      union { v16h v; h8 h[2]; } Bh, Bl;
      Bh.h[0] = *(const h8*)(Wh + rb);
      Bh.h[1] = *(const h8*)(Wh + rb + 16);
      Bl.h[0] = *(const h8*)(Wl + rb);
      Bl.h[1] = *(const h8*)(Wl + rb + 16);

      // f16 split-accumulate: A*B ~= Ah*Bh + Ah*Bl + Al*Bh  (Al*Bl ~ 2^-22, dropped)
      acc[j] = __builtin_amdgcn_wmma_f32_16x16x32_f16(false, Ah, false, Bl.v,
                                                      (short)0, acc[j], false, false);
      acc[j] = __builtin_amdgcn_wmma_f32_16x16x32_f16(false, Al, false, Bh.v,
                                                      (short)0, acc[j], false, false);
      acc[j] = __builtin_amdgcn_wmma_f32_16x16x32_f16(false, Ah, false, Bh.v,
                                                      (short)0, acc[j], false, false);
    }
  }

  // ---- Epilogue: C/D layout -> VGPR r = row (m0 + wv*16 + hi*8 + r), col = lr
#pragma unroll
  for (int j = 0; j < 4; ++j) {
    const int n = nbase + j * 16 + lr;
#pragma unroll
    for (int r = 0; r < 8; ++r) {
      int row = m0 + wv * 16 + hi * 8 + r;
      if (row < M) out[((size_t)row << 9) + n] = acc[j][r];
    }
  }
}

// ---------------------------------------------------------------------------
extern "C" void kernel_launch(void* const* d_in, const int* in_sizes, int n_in,
                              void* d_out, int out_size, void* d_ws, size_t ws_size,
                              hipStream_t stream) {
  const float* src_h  = (const float*)d_in[0];
  const float* W      = (const float*)d_in[1];
  const float* b      = (const float*)d_in[2];
  const int*   src_idx = (const int*)d_in[3];
  const int*   dst_idx = (const int*)d_in[4];
  const int n_edges = in_sizes[3];
  const int M = out_size / D_K;          // n_dst

  float* agg = (float*)d_ws;             // [M,512] f32, 102.4 MB (L2-resident)
  float* out = (float*)d_out;

  // Phase 0: zero agg
  int n4 = M * (D_K / 4);
  gnn_zero_kernel<<<(n4 + 255) / 256, 256, 0, stream>>>((float4*)agg, n4);

  // Phase 1: scatter-add, one thread per (edge, 4-float chunk)
  long total = (long)n_edges * (D_K / 4);
  gnn_scatter_kernel<<<(int)((total + 255) / 256), 256, 0, stream>>>(
      src_h, src_idx, dst_idx, agg, n_edges);

  // Phase 2: WMMA GEMM with bias epilogue
  dim3 grid((M + TILE_M - 1) / TILE_M, D_K / TILE_N);
  size_t lds_bytes = (size_t)2 * TILE_N * LDS_STRIDE * 2;   // 133,120 B dynamic LDS
  gnn_gemm_kernel<<<grid, 256, lds_bytes, stream>>>(agg, W, b, out, M);
}
